// Attention_69157563400952
// MI455X (gfx1250) — matrix operand order repaired and checked by reading, not compiled, not run on hardware
//
#include <hip/hip_runtime.h>
#include <hip/hip_bf16.h>

typedef __attribute__((ext_vector_type(16))) __bf16 v16bf;
typedef __attribute__((ext_vector_type(8)))  float  v8f;

#define NQ 16384
#define SCALE_ATTN 0.17677669529663687f  // 1/sqrt(32)

// native RNE f32 -> bf16 (lowers to v_cvt_*bf16* on gfx1250)
__device__ __forceinline__ unsigned short f32_to_bf16(float f) {
  __bf16 h = (__bf16)f;
  return __builtin_bit_cast(unsigned short, h);
}
__device__ __forceinline__ unsigned int pack_bf16x2(float a, float b) {
  return ((unsigned int)f32_to_bf16(b) << 16) | (unsigned int)f32_to_bf16(a);
}

__device__ __forceinline__ v8f zero8() { v8f z = {}; return z; }

__device__ __forceinline__ v8f wmma_bf16(v16bf a, v16bf b, v8f c) {
  return __builtin_amdgcn_wmma_f32_16x16x32_bf16(false, a, false, b, (short)0, c, false, false);
}

// Async 16B copy global -> LDS (GLOBAL_LOAD_ASYNC_TO_LDS_B128, ASYNCcnt).
// Per ISA 10.2: generic LDS pointer low 32 bits == LDS byte offset.
__device__ __forceinline__ void async_ld16(void* lds_ptr, const void* gptr) {
  unsigned int loff = (unsigned int)(size_t)lds_ptr;
  asm volatile("global_load_async_to_lds_b128 %0, %1, off"
               :: "v"(loff), "v"(gptr) : "memory");
}
__device__ __forceinline__ void wait_async0() {
  asm volatile("s_wait_asynccnt 0x0" ::: "memory");
}

// A fragment 16x32 bf16: element (m,k) = base[m*ldk + k]
__device__ __forceinline__ v16bf load_a_frag(const unsigned short* base, int ldk, int lane) {
  const int hh = lane >> 4;
  const int m  = lane & 15;
  union { v16bf v; unsigned int u[8]; } r;
  const unsigned short* p = base + m * ldk;
#pragma unroll
  for (int j = 0; j < 8; ++j) {
    int k0 = 2 * j + 16 * hh;  // dword pair (k0,k0+1)
    r.u[j] = *(const unsigned int*)(p + k0);
  }
  return r.v;
}

// B fragment 32x16 bf16 from K-contiguous storage: element (k,n) = base[n*ldk + k]
__device__ __forceinline__ v16bf load_bt_frag(const unsigned short* base, int ldk, int lane) {
  const int hh = lane >> 4;
  const int n  = lane & 15;
  union { v16bf v; unsigned int u[8]; } r;
  const unsigned short* p = base + n * ldk + 16 * hh;
#pragma unroll
  for (int j = 0; j < 8; ++j) r.u[j] = *(const unsigned int*)(p + 2 * j);
  return r.v;
}

// C/D f32 16x16: VGPR r -> rows r (lanes 0-15) / r+8 (lanes 16-31)
__device__ __forceinline__ void store_d_f32(float* base, int ldn, int lane, v8f d) {
  const int hh = lane >> 4;
  const int n  = lane & 15;
#pragma unroll
  for (int rr = 0; rr < 8; ++rr) base[(rr + 8 * hh) * ldn + n] = d[rr];
}

__device__ __forceinline__ void store_d_bf16(unsigned short* base, int ldn, int lane, v8f d) {
  const int hh = lane >> 4;
  const int n  = lane & 15;
#pragma unroll
  for (int rr = 0; rr < 8; ++rr) base[(rr + 8 * hh) * ldn + n] = f32_to_bf16(d[rr]);
}

// ---------------------------------------------------------------------------
// Kernel 0: weight prep (bf16 + transpose so every B operand is K-contiguous).
// ---------------------------------------------------------------------------
__global__ __launch_bounds__(256) void prep_kernel(
    const float* __restrict__ Wq, const float* __restrict__ Wkv,
    const float* __restrict__ Wsr, const float* __restrict__ Wproj,
    unsigned short* __restrict__ wq_t, unsigned short* __restrict__ wkv_t,
    unsigned short* __restrict__ wsr_t, unsigned short* __restrict__ wp_t) {
  int idx = blockIdx.x * 256 + threadIdx.x;
  for (; idx < 278528; idx += gridDim.x * 256) {
    if (idx < 262144) {
      int i = idx & 63; int rem = idx >> 6; int o = rem & 63; int s = rem >> 6;
      wsr_t[idx] = f32_to_bf16(Wsr[(o * 64 + i) * 64 + s]);
    } else if (idx < 262144 + 4096) {
      int j = idx - 262144; int n = j >> 6, k = j & 63;
      wq_t[j] = f32_to_bf16(Wq[k * 64 + n]);
    } else if (idx < 262144 + 4096 + 8192) {
      int j = idx - 262144 - 4096; int n = j >> 6, k = j & 63;
      wkv_t[j] = f32_to_bf16(Wkv[k * 128 + n]);
    } else {
      int j = idx - 262144 - 4096 - 8192; int n = j >> 6, k = j & 63;
      wp_t[j] = f32_to_bf16(Wproj[k * 64 + n]);
    }
  }
}

// ---------------------------------------------------------------------------
// Kernel 1: patch-embed conv (GEMM [64tok,4096]x[4096,64]) + b_sr -> LayerNorm
//           -> KV GEMM [64,64]x[64,128] -> K natural [b][h][nk][d],
//              V transposed [b][h][d][nk] (both B-fragment-ready, bf16).
// ---------------------------------------------------------------------------
__global__ __launch_bounds__(256) void kv_kernel(
    const float* __restrict__ x,
    const unsigned short* __restrict__ wsr_t,
    const float* __restrict__ b_sr,
    const float* __restrict__ ln_g,
    const float* __restrict__ ln_b,
    const unsigned short* __restrict__ wkv_t,
    unsigned short* __restrict__ Kt,   // [b][h][nk=256][d=32]
    unsigned short* __restrict__ Vt) { // [b][h][d=32][nk=256]
  __shared__ __align__(16) unsigned short sA[64 * 72];      // x chunk  [tok][i]
  __shared__ __align__(16) unsigned short sB[64 * 72];      // W chunk  [o][i]
  __shared__ __align__(16) unsigned short sWkv[128 * 72];   // [n=128][k=64]
  __shared__ __align__(16) float          sConv[64 * 68];
  __shared__ __align__(16) unsigned short sLN[64 * 72];
  __shared__ __align__(16) float          sKV[64 * 128];

  const int tid  = threadIdx.x;
  const int lane = tid & 31;
  const int w    = tid >> 5;
  const int b    = blockIdx.y;
  const int tile = blockIdx.x;

  // stage Wkv^T via async LDS loads (rows padded 64 -> 72)
#pragma unroll
  for (int i = 0; i < 4; ++i) {
    int c = tid + 256 * i;             // 1024 chunks of 16B
    int row = c >> 3, off = c & 7;
    async_ld16(sWkv + row * 72 + off * 8, wkv_t + row * 64 + off * 8);
  }

  const int r  = w & 3;          // row tile of 64x64 output
  const int cb = (w >> 2) * 2;   // 2 col tiles per wave
  v8f acc0 = zero8(), acc1 = zero8();

  const int tok = tid >> 2;      // A staging role
  const int seg = tid & 3;
  const int p   = tile * 64 + tok;
  const int ph  = p >> 4, pw = p & 15;

  for (int s = 0; s < 64; ++s) {  // s = kh*8+kw
    {
      int kh = s >> 3, kw = s & 7;
      int grow = (ph * 8 + kh) * 128 + pw * 8 + kw;
      const float4* src = (const float4*)(x + ((size_t)(b * NQ + grow)) * 64 + seg * 16);
      uint4* dst = (uint4*)(sA + tok * 72 + seg * 16);
      float4 f0 = src[0], f1 = src[1], f2 = src[2], f3 = src[3];
      uint4 p0, p1;
      p0.x = pack_bf16x2(f0.x, f0.y); p0.y = pack_bf16x2(f0.z, f0.w);
      p0.z = pack_bf16x2(f1.x, f1.y); p0.w = pack_bf16x2(f1.z, f1.w);
      p1.x = pack_bf16x2(f2.x, f2.y); p1.y = pack_bf16x2(f2.z, f2.w);
      p1.z = pack_bf16x2(f3.x, f3.y); p1.w = pack_bf16x2(f3.z, f3.w);
      dst[0] = p0; dst[1] = p1;
      // async stage weight chunk [64 o][64 i] (rows padded to 72)
#pragma unroll
      for (int i = 0; i < 2; ++i) {
        int c = tid + 256 * i;         // 512 chunks
        int row = c >> 3, off = c & 7;
        async_ld16(sB + row * 72 + off * 8, wsr_t + s * 4096 + row * 64 + off * 8);
      }
    }
    wait_async0();
    __syncthreads();
#pragma unroll
    for (int ks = 0; ks < 2; ++ks) {
      v16bf a  = load_a_frag (sA + r * 16 * 72 + ks * 32, 72, lane);
      v16bf b0 = load_bt_frag(sB + (cb    ) * 16 * 72 + ks * 32, 72, lane);
      v16bf b1 = load_bt_frag(sB + (cb + 1) * 16 * 72 + ks * 32, 72, lane);
      acc0 = wmma_bf16(a, b0, acc0);
      acc1 = wmma_bf16(a, b1, acc1);
    }
    __syncthreads();
  }
  store_d_f32(sConv + r * 16 * 68 + cb * 16, 68, lane, acc0);
  store_d_f32(sConv + r * 16 * 68 + (cb + 1) * 16, 68, lane, acc1);
  __syncthreads();

  // LayerNorm per token (channel dim 64), bias b_sr folded in
  if (tid < 64) {
    const float* cr = sConv + tid * 68;
    float m = 0.f;
    for (int o = 0; o < 64; ++o) m += cr[o] + b_sr[o];
    m *= (1.0f / 64.0f);
    float var = 0.f;
    for (int o = 0; o < 64; ++o) { float dv = cr[o] + b_sr[o] - m; var += dv * dv; }
    var *= (1.0f / 64.0f);
    float rstd = __frsqrt_rn(var + 1e-5f);
    unsigned int* lr = (unsigned int*)(sLN + tid * 72);
    for (int o = 0; o < 64; o += 2) {
      float y0 = (cr[o]     + b_sr[o]     - m) * rstd * ln_g[o]     + ln_b[o];
      float y1 = (cr[o + 1] + b_sr[o + 1] - m) * rstd * ln_g[o + 1] + ln_b[o + 1];
      lr[o >> 1] = pack_bf16x2(y0, y1);
    }
  }
  __syncthreads();

  // KV GEMM: [64,64] x [64,128]
  {
    int cb2 = (w >> 2) * 4;
    v8f acc[4];
#pragma unroll
    for (int i = 0; i < 4; ++i) acc[i] = zero8();
#pragma unroll
    for (int ks = 0; ks < 2; ++ks) {
      v16bf a = load_a_frag(sLN + r * 16 * 72 + ks * 32, 72, lane);
#pragma unroll
      for (int cc = 0; cc < 4; ++cc) {
        v16bf bb = load_bt_frag(sWkv + (cb2 + cc) * 16 * 72 + ks * 32, 72, lane);
        acc[cc] = wmma_bf16(a, bb, acc[cc]);
      }
    }
#pragma unroll
    for (int cc = 0; cc < 4; ++cc)
      store_d_f32(sKV + r * 16 * 128 + (cb2 + cc) * 16, 128, lane, acc[cc]);
  }
  __syncthreads();

  // scatter: K -> [b][h][nk][d] (natural), V -> [b][h][d][nk] (transposed)
  for (int i = tid; i < 64 * 128; i += 256) {
    int tk = i >> 7, col = i & 127;
    int nk = tile * 64 + tk;
    unsigned short bv = f32_to_bf16(sKV[i]);
    if (col < 64) {
      int hh = col >> 5, d0 = col & 31;
      Kt[((b * 2 + hh) * 256 + nk) * 32 + d0] = bv;
    } else {
      int c2 = col - 64;
      int hh = c2 >> 5, d0 = c2 & 31;
      Vt[((b * 2 + hh) * 32 + d0) * 256 + nk] = bv;
    }
  }
}

// ---------------------------------------------------------------------------
// Kernel 2: fused attention. 32 queries/block, K/V LDS-resident (async-staged).
// ---------------------------------------------------------------------------
__global__ __launch_bounds__(256) void attn_kernel(
    const float* __restrict__ x,
    const unsigned short* __restrict__ wq_t,
    const unsigned short* __restrict__ wp_t,
    const unsigned short* __restrict__ Kt,
    const unsigned short* __restrict__ Vt,
    const float* __restrict__ b_proj,
    float* __restrict__ out) {
  __shared__ __align__(16) unsigned short sK [2 * 256 * 40]; // [h][nk][d] pad 32->40
  __shared__ __align__(16) unsigned short sVt[2 * 32 * 264]; // [h][d][nk] pad 256->264
  __shared__ __align__(16) unsigned short sX [32 * 72];
  __shared__ __align__(16) unsigned short sQ [32 * 72];
  __shared__ __align__(16) float          sS [32 * 260];
  __shared__ __align__(16) unsigned short sP [2 * 32 * 264]; // unnormalized exp
  __shared__ __align__(16) unsigned short sO [32 * 72];
  __shared__ float sRed[32 * 4];
  __shared__ float sRowMax[32];
  __shared__ float sRinv[2 * 32];

  const int tid  = threadIdx.x;
  const int lane = tid & 31;
  const int w    = tid >> 5;
  const int b    = blockIdx.y;
  const int q0   = blockIdx.x * 32;

  {  // async-stage K (512 rows x 32) and V (64 rows x 256) with padded strides
    const unsigned short* gK = Kt + (size_t)b * 16384;
    const unsigned short* gV = Vt + (size_t)b * 16384;
#pragma unroll
    for (int i = 0; i < 8; ++i) {
      int c = tid + 256 * i;                       // 2048 chunks each
      int kr = c >> 2, ko = c & 3;                 // K: 4 chunks / row
      async_ld16(sK + kr * 40 + ko * 8, gK + kr * 32 + ko * 8);
      int vr = c >> 5, vo = c & 31;                // V: 32 chunks / row
      async_ld16(sVt + vr * 264 + vo * 8, gV + vr * 256 + vo * 8);
    }
  }
  {  // stage x tile -> bf16 (packed cvt, single b128 LDS store)
    int row = tid >> 3, seg = tid & 7;
    const float4* src = (const float4*)(x + ((size_t)(b * NQ + q0 + row)) * 64 + seg * 8);
    float4 f0 = src[0], f1 = src[1];
    uint4 pk;
    pk.x = pack_bf16x2(f0.x, f0.y); pk.y = pack_bf16x2(f0.z, f0.w);
    pk.z = pack_bf16x2(f1.x, f1.y); pk.w = pack_bf16x2(f1.z, f1.w);
    *(uint4*)(sX + row * 72 + seg * 8) = pk;
  }
  wait_async0();
  __syncthreads();

  // Q = X @ Wq  (transposed B read straight from global, L2-hot)
  {
    int r = w & 1, c = w >> 1;
    v8f acc = zero8();
#pragma unroll
    for (int ks = 0; ks < 2; ++ks) {
      v16bf a  = load_a_frag (sX + r * 16 * 72 + ks * 32, 72, lane);
      v16bf bb = load_bt_frag(wq_t + (c * 16) * 64 + ks * 32, 64, lane);
      acc = wmma_bf16(a, bb, acc);
    }
    store_d_bf16(sQ + r * 16 * 72 + c * 16, 72, lane, acc);
  }
  __syncthreads();

  for (int h = 0; h < 2; ++h) {
    {  // scores: [32 q][256 nk], dh=32 => one WMMA per 16x16 tile
      int r = w & 1, cb = (w >> 1) * 4;
      v16bf a = load_a_frag(sQ + r * 16 * 72 + h * 32, 72, lane);
#pragma unroll
      for (int cc = 0; cc < 4; ++cc) {
        v16bf bb = load_bt_frag(sK + h * (256 * 40) + (cb + cc) * 16 * 40, 40, lane);
        v8f acc = wmma_bf16(a, bb, zero8());
        store_d_f32(sS + r * 16 * 260 + (cb + cc) * 16, 260, lane, acc);
      }
    }
    __syncthreads();
    // softmax over 256 keys (4 threads per row, float4-vectorized)
    int row = tid >> 2, q4 = tid & 3;
    if (tid < 128) {
      const float4* sr4 = (const float4*)(sS + row * 260 + q4 * 64);
      float mx = -3.0e38f;
#pragma unroll
      for (int j = 0; j < 16; ++j) {
        float4 f = sr4[j];
        mx = fmaxf(mx, fmaxf(fmaxf(f.x, f.y), fmaxf(f.z, f.w)));
      }
      sRed[row * 4 + q4] = mx;
    }
    __syncthreads();
    if (tid < 32)
      sRowMax[tid] = fmaxf(fmaxf(sRed[tid * 4], sRed[tid * 4 + 1]),
                           fmaxf(sRed[tid * 4 + 2], sRed[tid * 4 + 3]));
    __syncthreads();
    if (tid < 128) {
      float mx = sRowMax[row];
      const float4* sr4 = (const float4*)(sS + row * 260 + q4 * 64);
      uint2* pp = (uint2*)(sP + h * 8448 + row * 264 + q4 * 64);
      float sum = 0.f;
#pragma unroll
      for (int j = 0; j < 16; ++j) {
        float4 f = sr4[j];
        float e0 = __expf((f.x - mx) * SCALE_ATTN);
        float e1 = __expf((f.y - mx) * SCALE_ATTN);
        float e2 = __expf((f.z - mx) * SCALE_ATTN);
        float e3 = __expf((f.w - mx) * SCALE_ATTN);
        sum += (e0 + e1) + (e2 + e3);
        uint2 pk;
        pk.x = pack_bf16x2(e0, e1);
        pk.y = pack_bf16x2(e2, e3);
        pp[j] = pk;
      }
      sRed[row * 4 + q4] = sum;
    }
    __syncthreads();
    if (tid < 32)
      sRinv[h * 32 + tid] = 1.0f / (sRed[tid * 4] + sRed[tid * 4 + 1] +
                                    sRed[tid * 4 + 2] + sRed[tid * 4 + 3]);
    __syncthreads();
  }

  // P @ V, both heads in parallel across 8 waves; 1/rowsum folded into store
  {
    int hh = w >> 2, r = (w >> 1) & 1, c = w & 1;
    v8f acc = zero8();
#pragma unroll
    for (int ks = 0; ks < 8; ++ks) {
      v16bf a  = load_a_frag (sP + hh * 8448 + r * 16 * 264 + ks * 32, 264, lane);
      v16bf bb = load_bt_frag(sVt + hh * (32 * 264) + (c * 16) * 264 + ks * 32, 264, lane);
      acc = wmma_bf16(a, bb, acc);
    }
    const float* rinv = sRinv + hh * 32;
    int lh = lane >> 4, n = lane & 15;
    unsigned short* base = sO + (r * 16) * 72 + hh * 32 + c * 16;
#pragma unroll
    for (int rr = 0; rr < 8; ++rr) {
      int m = rr + 8 * lh;
      base[m * 72 + n] = f32_to_bf16(acc[rr] * rinv[r * 16 + m]);
    }
  }
  __syncthreads();

  // out = O @ Wproj + b_proj
  {
    int r = w & 1, c = w >> 1;
    v8f acc = zero8();
#pragma unroll
    for (int ks = 0; ks < 2; ++ks) {
      v16bf a  = load_a_frag (sO + r * 16 * 72 + ks * 32, 72, lane);
      v16bf bb = load_bt_frag(wp_t + (c * 16) * 64 + ks * 32, 64, lane);
      acc = wmma_bf16(a, bb, acc);
    }
    int lh = lane >> 4, n = lane & 15;
#pragma unroll
    for (int rr = 0; rr < 8; ++rr) {
      int m = q0 + r * 16 + rr + 8 * lh;
      int col = c * 16 + n;
      out[((size_t)(b * NQ + m)) * 64 + col] = acc[rr] + b_proj[col];
    }
  }
}

extern "C" void kernel_launch(void* const* d_in, const int* in_sizes, int n_in,
                              void* d_out, int out_size, void* d_ws, size_t ws_size,
                              hipStream_t stream) {
  (void)in_sizes; (void)n_in; (void)out_size; (void)ws_size;
  const float* x      = (const float*)d_in[0];
  // d_in[1]=height, d_in[2]=width (fixed 128x128)
  const float* Wq     = (const float*)d_in[3];
  const float* Wkv    = (const float*)d_in[4];
  const float* Wsr    = (const float*)d_in[5];
  const float* b_sr   = (const float*)d_in[6];
  const float* ln_g   = (const float*)d_in[7];
  const float* ln_b   = (const float*)d_in[8];
  const float* Wproj  = (const float*)d_in[9];
  const float* b_proj = (const float*)d_in[10];
  float* out = (float*)d_out;

  unsigned char* ws = (unsigned char*)d_ws;
  unsigned short* wq_t  = (unsigned short*)(ws);                       //   8 KB
  unsigned short* wkv_t = (unsigned short*)(ws + 8192);                //  16 KB
  unsigned short* wp_t  = (unsigned short*)(ws + 24576);               //   8 KB
  unsigned short* wsr_t = (unsigned short*)(ws + 32768);               // 512 KB
  unsigned short* Kt    = (unsigned short*)(ws + 32768 + 524288);      // 256 KB
  unsigned short* Vt    = (unsigned short*)(ws + 32768 + 524288 + 262144); // 256 KB

  prep_kernel<<<dim3(1088), dim3(256), 0, stream>>>(Wq, Wkv, Wsr, Wproj,
                                                    wq_t, wkv_t, wsr_t, wp_t);
  kv_kernel<<<dim3(4, 8), dim3(256), 0, stream>>>(x, wsr_t, b_sr, ln_g, ln_b,
                                                  wkv_t, Kt, Vt);
  attn_kernel<<<dim3(512, 8), dim3(256), 0, stream>>>(x, wq_t, wp_t, Kt, Vt,
                                                      b_proj, out);
}
